// gcn_14388140441617
// MI455X (gfx1250) — matrix operand order repaired and checked by reading, not compiled, not run on hardware
//
#include <hip/hip_runtime.h>
#include <hip/hip_bf16.h>

typedef __attribute__((ext_vector_type(16))) __bf16 v16bf;
typedef __attribute__((ext_vector_type(2)))  __bf16 v2bf;
typedef __attribute__((ext_vector_type(8)))  float  v8f;

constexpr int NB = 64;    // batches
constexpr int NN = 512;   // nodes
constexpr int H1 = 256;
constexpr int H2 = 128;
constexpr int MB = 64;    // rows per workgroup
constexpr int KT = 32;    // K per WMMA step (bf16 shape 16x16x32)

#if __has_builtin(__builtin_amdgcn_cvt_pk_bf16_f32)
static __device__ __forceinline__ unsigned pack2(float a, float b) {
  v2bf r = __builtin_amdgcn_cvt_pk_bf16_f32(a, b);
  return __builtin_bit_cast(unsigned, r);
}
static __device__ __forceinline__ __bf16 f2bf(float f) {
  v2bf r = __builtin_amdgcn_cvt_pk_bf16_f32(f, 0.0f);
  return r[0];
}
#else
static __device__ __forceinline__ __bf16 f2bf(float f) {
  unsigned u = __builtin_bit_cast(unsigned, f);
  u = (u + 0x7FFFu + ((u >> 16) & 1u)) >> 16;
  unsigned short h = (unsigned short)u;
  return __builtin_bit_cast(__bf16, h);
}
static __device__ __forceinline__ unsigned pack2(float a, float b) {
  unsigned x = (unsigned)__builtin_bit_cast(unsigned short, f2bf(a));
  unsigned y = (unsigned)__builtin_bit_cast(unsigned short, f2bf(b));
  return x | (y << 16);
}
#endif

static __device__ __forceinline__ unsigned short bf_bits(__bf16 v) {
  return __builtin_bit_cast(unsigned short, v);
}
static __device__ __forceinline__ float relu(float x) {
  return __builtin_fmaxf(x, 0.0f);
}

union Frag { v16bf v; uint4 q[2]; };

__global__ void zero_out(float* out, int n) {
  int i = blockIdx.x * 256 + threadIdx.x;
  if (i < n) out[i] = 0.0f;
}

// W2T_bf[h2][h1] = bf16(W2[h1][h2])  (one-time, 64 KB)
__global__ __launch_bounds__(256) void w2_transpose(
    const float* __restrict__ W2, __bf16* __restrict__ W2T) {
  int col = blockIdx.x;   // h2: 0..127
  int k   = threadIdx.x;  // h1: 0..255
  W2T[(size_t)col * H1 + k] = f2bf(W2[(size_t)k * H2 + col]);
}

// H0T[b][h1][node] = bf16(weight[X_idx[b,node], h1] + bias[h1])
// Gather coalesced along h1, transpose through LDS, store coalesced along node.
__global__ __launch_bounds__(256) void gather_embed_t(
    const int* __restrict__ Xidx, const float* __restrict__ W,
    const float* __restrict__ bias, __bf16* __restrict__ H0T) {
  const int b = blockIdx.y, n0 = blockIdx.x * 32;
  const int tid = threadIdx.x;
  __shared__ int idxs[32];
  __shared__ alignas(16) __bf16 Tr[32][264];

  if (tid < 32) idxs[tid] = Xidx[b * NN + n0 + tid];
  __syncthreads();
  // 32 nodes x 256 cols = 2048 float4, 8 per thread
  #pragma unroll
  for (int i = 0; i < 8; ++i) {
    int p = tid + i * 256;
    int node = p >> 6, c4 = (p & 63) << 2;
    const float4 f = *(const float4*)(W + (size_t)idxs[node] * H1 + c4);
    uint2 pk;
    pk.x = pack2(f.x + bias[c4 + 0], f.y + bias[c4 + 1]);
    pk.y = pack2(f.z + bias[c4 + 2], f.w + bias[c4 + 3]);
    *(uint2*)&Tr[node][c4] = pk;
  }
  __syncthreads();
  // transposed write: 256 cols x 32 nodes = 1024 uint4, 4 per thread
  #pragma unroll
  for (int i = 0; i < 4; ++i) {
    int w = tid + i * 256;
    int c = w >> 2, nd0 = (w & 3) << 3;
    unsigned e0 = (unsigned)bf_bits(Tr[nd0 + 0][c]) | ((unsigned)bf_bits(Tr[nd0 + 1][c]) << 16);
    unsigned e1 = (unsigned)bf_bits(Tr[nd0 + 2][c]) | ((unsigned)bf_bits(Tr[nd0 + 3][c]) << 16);
    unsigned e2 = (unsigned)bf_bits(Tr[nd0 + 4][c]) | ((unsigned)bf_bits(Tr[nd0 + 5][c]) << 16);
    unsigned e3 = (unsigned)bf_bits(Tr[nd0 + 6][c]) | ((unsigned)bf_bits(Tr[nd0 + 7][c]) << 16);
    uint4 q; q.x = e0; q.y = e1; q.z = e2; q.w = e3;
    *(uint4*)(H0T + ((size_t)b * H1 + c) * NN + n0 + nd0) = q;
  }
}

// Per (batch, 64-row block): C1 = relu(A @ H0) [64x256] kept in regs/LDS,
// then T = C1 @ W2 + bias2 [64x128] -> TT[b][h2][node] bf16 (transposed).
__global__ __launch_bounds__(256) void gcn_layer1_fused(
    const float* __restrict__ A,       // [NB, NN, NN] f32
    const __bf16* __restrict__ H0T,    // [NB, H1, NN] bf16 (K contiguous)
    const __bf16* __restrict__ W2T,    // [H2, H1] bf16 (K contiguous)
    const float* __restrict__ bias2,   // [H2] f32
    __bf16* __restrict__ TT) {         // [NB, H2, NN] bf16
  const int b   = blockIdx.y;
  const int mb0 = blockIdx.x * MB;
  const int tid = threadIdx.x;
  const int wave = tid >> 5, lane = tid & 31;
  const int half = lane >> 4, ln = lane & 15;

  __shared__ alignas(16) __bf16 As[MB][40];      // A tile, rows 16B-aligned
  __shared__ alignas(16) __bf16 uBuf[256 * 40];  // overlay: BsT / WsT / Ttmp
  __shared__ alignas(16) __bf16 Cs[MB][264];     // relu(C1) slab

  auto BsT  = (__bf16(*)[40])uBuf;    // [256][40]  H0 tile, col-major (K contig)
  auto WsT  = (__bf16(*)[40])uBuf;    // [128][40]  W2 tile, col-major
  auto Ttmp = (__bf16(*)[132])uBuf;   // [64][132]  T transpose staging

  const float*  Ab   = A   + (size_t)b * NN * NN;
  const __bf16* H0Tb = H0T + (size_t)b * H1 * NN;

  // ---------------- Phase 1: C1 = relu(A @ H0) ----------------
  const int mr  = (wave & 3) * 16;
  const int nc0 = (wave >> 2) * 128;

  v8f acc[8];
  #pragma unroll
  for (int j = 0; j < 8; ++j)
    #pragma unroll
    for (int e = 0; e < 8; ++e) acc[j][e] = 0.0f;

  for (int kk = 0; kk < NN; kk += KT) {
    __syncthreads();
    // A tile 64x32 f32->bf16: 512 float4 reads, packed 8B LDS stores
    #pragma unroll
    for (int i = 0; i < 2; ++i) {
      int p = tid + i * 256;
      int row = p >> 3, c4 = (p & 7) << 2;
      const float4 f = *(const float4*)(Ab + (size_t)(mb0 + row) * NN + kk + c4);
      uint2 pk; pk.x = pack2(f.x, f.y); pk.y = pack2(f.z, f.w);
      *(uint2*)&As[row][c4] = pk;
    }
    // B tile: 256 cols x 32 K, straight b128 copy (already K-contiguous)
    #pragma unroll
    for (int i = 0; i < 4; ++i) {
      int w = tid + i * 256;
      int c = w >> 2, koff = (w & 3) << 3;
      *(uint4*)&BsT[c][koff] = *(const uint4*)(H0Tb + (size_t)c * NN + kk + koff);
    }
    __syncthreads();

    Frag af;
    af.q[0] = *(const uint4*)&As[mr + ln][half * 16];
    af.q[1] = *(const uint4*)&As[mr + ln][half * 16 + 8];
    #pragma unroll
    for (int j = 0; j < 8; ++j) {
      Frag bf;
      const __bf16* s = &BsT[nc0 + 16 * j + ln][half * 16];
      bf.q[0] = *(const uint4*)(s);
      bf.q[1] = *(const uint4*)(s + 8);
      acc[j] = __builtin_amdgcn_wmma_f32_16x16x32_bf16(
          false, af.v, false, bf.v, (short)0, acc[j], false, false);
    }
  }

  // relu -> Cs (C layout: m = v + 8*half, n = ln)
  #pragma unroll
  for (int j = 0; j < 8; ++j)
    #pragma unroll
    for (int v = 0; v < 8; ++v)
      Cs[mr + half * 8 + v][nc0 + 16 * j + ln] = f2bf(relu(acc[j][v]));

  // ---------------- Phase 2: T = relu(C1) @ W2 + bias2 ----------------
  const int mr2 = (wave & 3) * 16;
  const int nc2 = (wave >> 2) * 64;

  v8f acc2[4];
  #pragma unroll
  for (int j = 0; j < 4; ++j)
    #pragma unroll
    for (int e = 0; e < 8; ++e) acc2[j][e] = 0.0f;

  for (int kk = 0; kk < H1; kk += KT) {
    __syncthreads();  // orders Cs writes & previous WsT reads
    // W2 tile: 128 cols x 32 K: 512 uint4, 2 per thread
    #pragma unroll
    for (int i = 0; i < 2; ++i) {
      int w = tid + i * 256;
      int c = w >> 2, koff = (w & 3) << 3;
      *(uint4*)&WsT[c][koff] = *(const uint4*)(W2T + (size_t)c * H1 + kk + koff);
    }
    __syncthreads();

    Frag af;
    af.q[0] = *(const uint4*)&Cs[mr2 + ln][kk + half * 16];
    af.q[1] = *(const uint4*)&Cs[mr2 + ln][kk + half * 16 + 8];
    #pragma unroll
    for (int j = 0; j < 4; ++j) {
      Frag bf;
      const __bf16* s = &WsT[nc2 + 16 * j + ln][half * 16];
      bf.q[0] = *(const uint4*)(s);
      bf.q[1] = *(const uint4*)(s + 8);
      acc2[j] = __builtin_amdgcn_wmma_f32_16x16x32_bf16(
          false, af.v, false, bf.v, (short)0, acc2[j], false, false);
    }
  }

  __syncthreads();  // WsT reads done before Ttmp overlay writes
  #pragma unroll
  for (int j = 0; j < 4; ++j)
    #pragma unroll
    for (int v = 0; v < 8; ++v) {
      int col = nc2 + 16 * j + ln;
      Ttmp[mr2 + half * 8 + v][col] = f2bf(acc2[j][v] + bias2[col]);
    }
  __syncthreads();
  // transposed coalesced store: TT[b][col][mb0..mb0+63]; 1024 uint4, 4/thread
  __bf16* TTb = TT + (size_t)b * H2 * NN;
  #pragma unroll
  for (int i = 0; i < 4; ++i) {
    int w = tid + i * 256;
    int c = w >> 3, nd0 = (w & 7) << 3;
    unsigned e0 = (unsigned)bf_bits(Ttmp[nd0 + 0][c]) | ((unsigned)bf_bits(Ttmp[nd0 + 1][c]) << 16);
    unsigned e1 = (unsigned)bf_bits(Ttmp[nd0 + 2][c]) | ((unsigned)bf_bits(Ttmp[nd0 + 3][c]) << 16);
    unsigned e2 = (unsigned)bf_bits(Ttmp[nd0 + 4][c]) | ((unsigned)bf_bits(Ttmp[nd0 + 5][c]) << 16);
    unsigned e3 = (unsigned)bf_bits(Ttmp[nd0 + 6][c]) | ((unsigned)bf_bits(Ttmp[nd0 + 7][c]) << 16);
    uint4 q; q.x = e0; q.y = e1; q.z = e2; q.w = e3;
    *(uint4*)(TTb + (size_t)c * NN + mb0 + nd0) = q;
  }
}

// C2 = relu(A @ T), out[b] += column mean of this 64-row block
__global__ __launch_bounds__(256) void gcn_layer2_mean(
    const float* __restrict__ A,      // [NB, NN, NN] f32
    const __bf16* __restrict__ TT,    // [NB, H2, NN] bf16 (K contiguous)
    float* __restrict__ out) {        // [NB, H2] f32 (pre-zeroed)
  const int b   = blockIdx.y;
  const int mb0 = blockIdx.x * MB;
  const int tid = threadIdx.x;
  const int wave = tid >> 5, lane = tid & 31;
  const int half = lane >> 4, ln = lane & 15;

  __shared__ alignas(16) __bf16 As[MB][40];
  __shared__ alignas(16) __bf16 TsT[H2][40];
  __shared__ float colsum[H2];

  if (tid < H2) colsum[tid] = 0.0f;

  const float*  Ab  = A  + (size_t)b * NN * NN;
  const __bf16* TTb = TT + (size_t)b * H2 * NN;

  const int mr = (wave >> 1) * 16;
  const int nc = (wave & 1) * 64;

  v8f acc[4];
  #pragma unroll
  for (int j = 0; j < 4; ++j)
    #pragma unroll
    for (int e = 0; e < 8; ++e) acc[j][e] = 0.0f;

  for (int kk = 0; kk < NN; kk += KT) {
    __syncthreads();
    #pragma unroll
    for (int i = 0; i < 2; ++i) {
      int p = tid + i * 256;
      int row = p >> 3, c4 = (p & 7) << 2;
      const float4 f = *(const float4*)(Ab + (size_t)(mb0 + row) * NN + kk + c4);
      uint2 pk; pk.x = pack2(f.x, f.y); pk.y = pack2(f.z, f.w);
      *(uint2*)&As[row][c4] = pk;
    }
    // T tile: 128 cols x 32 K: 512 uint4, 2 per thread (b128 copy)
    #pragma unroll
    for (int i = 0; i < 2; ++i) {
      int w = tid + i * 256;
      int c = w >> 2, koff = (w & 3) << 3;
      *(uint4*)&TsT[c][koff] = *(const uint4*)(TTb + (size_t)c * NN + kk + koff);
    }
    __syncthreads();

    Frag af;
    af.q[0] = *(const uint4*)&As[mr + ln][half * 16];
    af.q[1] = *(const uint4*)&As[mr + ln][half * 16 + 8];
    #pragma unroll
    for (int j = 0; j < 4; ++j) {
      Frag bf;
      const __bf16* s = &TsT[nc + 16 * j + ln][half * 16];
      bf.q[0] = *(const uint4*)(s);
      bf.q[1] = *(const uint4*)(s + 8);
      acc[j] = __builtin_amdgcn_wmma_f32_16x16x32_bf16(
          false, af.v, false, bf.v, (short)0, acc[j], false, false);
    }
  }

  #pragma unroll
  for (int j = 0; j < 4; ++j) {
    float s = 0.0f;
    #pragma unroll
    for (int v = 0; v < 8; ++v) s += relu(acc[j][v]);
    atomicAdd(&colsum[nc + 16 * j + ln], s);
  }
  __syncthreads();
  if (tid < H2)
    atomicAdd(&out[b * H2 + tid], colsum[tid] * (1.0f / (float)NN));
}

extern "C" void kernel_launch(void* const* d_in, const int* in_sizes, int n_in,
                              void* d_out, int out_size, void* d_ws, size_t ws_size,
                              hipStream_t stream) {
  const float* A     = (const float*)d_in[0];
  const int*   Xidx  = (const int*)d_in[1];
  const float* W     = (const float*)d_in[2];
  const float* bias  = (const float*)d_in[3];
  const float* W2    = (const float*)d_in[4];
  const float* bias2 = (const float*)d_in[5];
  float* out = (float*)d_out;

  char* ws = (char*)d_ws;
  __bf16* H0T  = (__bf16*)ws;                                         // [64][256][512] = 16 MB
  __bf16* TT   = (__bf16*)(ws + (size_t)NB * H1 * NN * 2);            // [64][128][512] = 8 MB
  __bf16* W2Tb = (__bf16*)(ws + (size_t)NB * (H1 + H2) * NN * 2);     // [128][256] = 64 KB

  zero_out<<<(out_size + 255) / 256, 256, 0, stream>>>(out, out_size);
  w2_transpose<<<H2, 256, 0, stream>>>(W2, W2Tb);

  dim3 gg(NN / 32, NB);  // (16, 64)
  gather_embed_t<<<gg, 256, 0, stream>>>(Xidx, W, bias, H0T);

  dim3 grid(NN / MB, NB);  // (8, 64)
  gcn_layer1_fused<<<grid, 256, 0, stream>>>(A, H0T, W2Tb, bias2, TT);
  gcn_layer2_mean<<<grid, 256, 0, stream>>>(A, TT, out);
}
